// Attention_81149112090763
// MI455X (gfx1250) — matrix operand order repaired and checked by reading, not compiled, not run on hardware
//
#include <hip/hip_runtime.h>
#include <hip/hip_bf16.h>

// MI455X / gfx1250 causal MHA: bf16 WMMA (16x16x32, f32 accum) for all three
// matmul stages + per-wave online-softmax flash attention.

typedef __attribute__((ext_vector_type(16))) __bf16 v16bf;
typedef __attribute__((ext_vector_type(8)))  float  v8f;

union FragBF {
    v16bf v;
    uint4 q[2];
};

__device__ __forceinline__ unsigned short f2bf(float f) {
    unsigned u = __float_as_uint(f);
    unsigned r = u + 0x7FFFu + ((u >> 16) & 1u);   // RNE
    return (unsigned short)(r >> 16);
}

__device__ __forceinline__ v8f wmma_bf16(v16bf a, v16bf b, v8f c) {
    return __builtin_amdgcn_wmma_f32_16x16x32_bf16(
        /*neg_a=*/false, a, /*neg_b=*/false, b,
        /*c_mod=*/(short)0, c, /*reuse_a=*/false, /*reuse_b=*/false);
}

// ---------------------------------------------------------------- pack / transpose
__global__ void pack_x_kernel(const float* __restrict__ x,
                              unsigned short* __restrict__ xb, int n) {
    int i = blockIdx.x * 256 + threadIdx.x;
    if (i < n) xb[i] = f2bf(x[i]);
}

// w: [R][C] f32 row-major  ->  wT: [C][R] bf16 row-major
__global__ void transpose_w_kernel(const float* __restrict__ w,
                                   unsigned short* __restrict__ wT,
                                   int R, int C) {
    int i = blockIdx.x * 256 + threadIdx.x;
    if (i < R * C) {
        int c = i / R;
        int r = i - c * R;
        wT[i] = f2bf(w[r * C + c]);
    }
}

// ---------------------------------------------------------------- QKV projection
// xb: [4096][1024] bf16 ; wqT: [3072][1024] bf16 (transposed w_qkv)
// outputs: q,k : [32][2048][64] bf16 ; vt : [32][64][2048] bf16
__global__ __launch_bounds__(256) void qkv_gemm_kernel(
        const unsigned short* __restrict__ xb,
        const unsigned short* __restrict__ wqT,
        unsigned short* __restrict__ q,
        unsigned short* __restrict__ k,
        unsigned short* __restrict__ vt) {
    const int lane = threadIdx.x & 31;
    const int wave = threadIdx.x >> 5;
    const int half = lane >> 4;
    const int l16  = lane & 15;
    const int mbase = blockIdx.x * 128 + wave * 16;
    const int cbase = blockIdx.y * 64;

    v8f acc[4] = {};
    const unsigned short* aRow = xb + (size_t)(mbase + l16) * 1024 + half * 16;

    for (int k0 = 0; k0 < 1024; k0 += 32) {
        FragBF a;
        a.q[0] = *(const uint4*)(aRow + k0);        // K = 16*half + 0..7
        a.q[1] = *(const uint4*)(aRow + k0 + 8);    // K = 16*half + 8..15
#pragma unroll
        for (int t = 0; t < 4; ++t) {
            const unsigned short* bRow =
                wqT + (size_t)(cbase + 16 * t + l16) * 1024 + k0 + half * 16;
            FragBF b;
            b.q[0] = *(const uint4*)(bRow);         // K = 16*half + 0..7
            b.q[1] = *(const uint4*)(bRow + 8);     // K = 16*half + 8..15
            acc[t] = wmma_bf16(a.v, b.v, acc[t]);
        }
    }

#pragma unroll
    for (int t = 0; t < 4; ++t) {
        const int col   = cbase + 16 * t + l16;      // 0..3071
        const int which = col >> 10;                 // 0=q 1=k 2=v
        const int wi    = col & 1023;
        const int h     = wi >> 6;
        const int dd    = wi & 63;
#pragma unroll
        for (int r = 0; r < 8; ++r) {
            const int row = mbase + r + 8 * half;    // 0..4095
            const int b_  = row >> 11;
            const int n   = row & 2047;
            const int bh  = b_ * 16 + h;
            const unsigned short val = f2bf(acc[t][r]);
            if (which == 0)      q [((size_t)bh * 2048 + n) * 64 + dd] = val;
            else if (which == 1) k [((size_t)bh * 2048 + n) * 64 + dd] = val;
            else                 vt[((size_t)bh * 64 + dd) * 2048 + n] = val;
        }
    }
}

// ---------------------------------------------------------------- flash attention
// grid: x = N/128 query tiles, y = B*H. One wave = 16 query rows, d = 64.
__global__ __launch_bounds__(256) void flash_attn_kernel(
        const unsigned short* __restrict__ q,
        const unsigned short* __restrict__ k,
        const unsigned short* __restrict__ vt,
        unsigned short* __restrict__ attn) {
    __shared__ __align__(16) unsigned short plds[8][16][32];  // per-wave P bounce

    const int lane = threadIdx.x & 31;
    const int wave = threadIdx.x >> 5;
    const int half = lane >> 4;
    const int l16  = lane & 15;

    const int bh    = blockIdx.y;          // 0..31
    const int b_    = bh >> 4;
    const int h     = bh & 15;
    const int qrow0 = blockIdx.x * 128 + wave * 16;
    const float scale = 0.125f;            // 1/sqrt(64)

    const unsigned short* qbh = q  + (size_t)bh * 2048 * 64;
    const unsigned short* kbh = k  + (size_t)bh * 2048 * 64;
    const unsigned short* vbh = vt + (size_t)bh * 64 * 2048;

    // Q fragments: 16 rows x 64, two K-steps of 32
    FragBF qa[2];
    {
        const unsigned short* qr = qbh + (size_t)(qrow0 + l16) * 64 + half * 16;
        qa[0].q[0] = *(const uint4*)(qr);
        qa[0].q[1] = *(const uint4*)(qr + 8);
        qa[1].q[0] = *(const uint4*)(qr + 32);
        qa[1].q[1] = *(const uint4*)(qr + 40);
    }

    v8f o[4] = {};
    float mrow[8], lrow[8];
#pragma unroll
    for (int r = 0; r < 8; ++r) { mrow[r] = -__builtin_inff(); lrow[r] = 0.f; }

    const int ntiles = (qrow0 + 16 + 31) >> 5;   // causal: keys <= qrow0+15

    for (int jt = 0; jt < ntiles; ++jt) {
        const int j0 = jt * 32;

        // ---- S = Q K^T  (two 16x16 n-tiles, K=64)
        v8f s[2] = {};
#pragma unroll
        for (int nt = 0; nt < 2; ++nt) {
            const unsigned short* kr =
                kbh + (size_t)(j0 + nt * 16 + l16) * 64 + half * 16;
            FragBF kb0, kb1;
            kb0.q[0] = *(const uint4*)(kr);
            kb0.q[1] = *(const uint4*)(kr + 8);
            kb1.q[0] = *(const uint4*)(kr + 32);
            kb1.q[1] = *(const uint4*)(kr + 40);
            s[nt] = wmma_bf16(qa[0].v, kb0.v, s[nt]);
            s[nt] = wmma_bf16(qa[1].v, kb1.v, s[nt]);
        }

        // ---- scale + causal mask + online softmax stats
#pragma unroll
        for (int r = 0; r < 8; ++r) {
            const int row = qrow0 + r + 8 * half;
            float best = -__builtin_inff();
#pragma unroll
            for (int nt = 0; nt < 2; ++nt) {
                const int col = j0 + nt * 16 + l16;
                float sv = s[nt][r] * scale;
                sv = (col > row) ? -__builtin_inff() : sv;
                s[nt][r] = sv;
                best = fmaxf(best, sv);
            }
            best = fmaxf(best, __shfl_xor(best, 1));
            best = fmaxf(best, __shfl_xor(best, 2));
            best = fmaxf(best, __shfl_xor(best, 4));
            best = fmaxf(best, __shfl_xor(best, 8));

            const float mnew  = fmaxf(mrow[r], best);
            const float alpha = __expf(mrow[r] - mnew);
            mrow[r] = mnew;

            float rsum = 0.f;
#pragma unroll
            for (int nt = 0; nt < 2; ++nt) {
                const float p = __expf(s[nt][r] - mnew);
                s[nt][r] = p;
                rsum += p;
            }
            rsum += __shfl_xor(rsum, 1);
            rsum += __shfl_xor(rsum, 2);
            rsum += __shfl_xor(rsum, 4);
            rsum += __shfl_xor(rsum, 8);
            lrow[r] = lrow[r] * alpha + rsum;
#pragma unroll
            for (int t = 0; t < 4; ++t) o[t][r] *= alpha;
        }

        // ---- transpose P (C-layout) -> A-fragment layout via per-wave LDS
#pragma unroll
        for (int nt = 0; nt < 2; ++nt)
#pragma unroll
            for (int r = 0; r < 8; ++r)
                plds[wave][r + 8 * half][nt * 16 + l16] = f2bf(s[nt][r]);

        asm volatile("s_wait_dscnt 0x0" ::: "memory");

        FragBF pa;
        {
            const unsigned short* pr = &plds[wave][l16][half * 16];
            pa.q[0] = *(const uint4*)(pr);
            pa.q[1] = *(const uint4*)(pr + 8);
        }

        // ---- O += P V   (vT: [d][N], B-fragment contiguous along keys)
#pragma unroll
        for (int t = 0; t < 4; ++t) {
            const unsigned short* vr =
                vbh + (size_t)(t * 16 + l16) * 2048 + j0 + half * 16;
            FragBF vb;
            vb.q[0] = *(const uint4*)(vr);
            vb.q[1] = *(const uint4*)(vr + 8);
            o[t] = wmma_bf16(pa.v, vb.v, o[t]);
        }
    }

    // ---- normalize + store attn output as bf16 [B][N][H*64]
#pragma unroll
    for (int t = 0; t < 4; ++t) {
#pragma unroll
        for (int r = 0; r < 8; ++r) {
            const int row = qrow0 + r + 8 * half;
            const int dd  = t * 16 + l16;
            const float val = o[t][r] / lrow[r];
            attn[((size_t)b_ * 2048 + row) * 1024 + h * 64 + dd] = f2bf(val);
        }
    }
}

// ---------------------------------------------------------------- output projection
// ab: [4096][1024] bf16 ; woT: [1024][1024] bf16 (transposed) ; out: f32
__global__ __launch_bounds__(256) void out_gemm_kernel(
        const unsigned short* __restrict__ ab,
        const unsigned short* __restrict__ woT,
        float* __restrict__ out) {
    const int lane = threadIdx.x & 31;
    const int wave = threadIdx.x >> 5;
    const int half = lane >> 4;
    const int l16  = lane & 15;
    const int mbase = blockIdx.x * 128 + wave * 16;
    const int cbase = blockIdx.y * 64;

    v8f acc[4] = {};
    const unsigned short* aRow = ab + (size_t)(mbase + l16) * 1024 + half * 16;

    for (int k0 = 0; k0 < 1024; k0 += 32) {
        FragBF a;
        a.q[0] = *(const uint4*)(aRow + k0);
        a.q[1] = *(const uint4*)(aRow + k0 + 8);
#pragma unroll
        for (int t = 0; t < 4; ++t) {
            const unsigned short* bRow =
                woT + (size_t)(cbase + 16 * t + l16) * 1024 + k0 + half * 16;
            FragBF b;
            b.q[0] = *(const uint4*)(bRow);
            b.q[1] = *(const uint4*)(bRow + 8);
            acc[t] = wmma_bf16(a.v, b.v, acc[t]);
        }
    }

#pragma unroll
    for (int t = 0; t < 4; ++t) {
        const int col = cbase + 16 * t + l16;
#pragma unroll
        for (int r = 0; r < 8; ++r) {
            const int row = mbase + r + 8 * half;
            out[(size_t)row * 1024 + col] = acc[t][r];
        }
    }
}

// ---------------------------------------------------------------- launch
extern "C" void kernel_launch(void* const* d_in, const int* in_sizes, int n_in,
                              void* d_out, int out_size, void* d_ws, size_t ws_size,
                              hipStream_t stream) {
    const float* x     = (const float*)d_in[0];   // [2,2048,1024]
    const float* w_qkv = (const float*)d_in[1];   // [1024,3072]
    const float* w_out = (const float*)d_in[2];   // [1024,1024]
    float* out = (float*)d_out;                   // [2,2048,1024]

    char* ws = (char*)d_ws;
    size_t off = 0;
    auto carve = [&](size_t bytes) -> void* {
        void* p = ws + off;
        off += (bytes + 255) & ~(size_t)255;
        return p;
    };

    unsigned short* xb  = (unsigned short*)carve((size_t)4096 * 1024 * 2);
    unsigned short* wqT = (unsigned short*)carve((size_t)3072 * 1024 * 2);
    unsigned short* woT = (unsigned short*)carve((size_t)1024 * 1024 * 2);
    unsigned short* qb  = (unsigned short*)carve((size_t)32 * 2048 * 64 * 2);
    unsigned short* kb  = (unsigned short*)carve((size_t)32 * 2048 * 64 * 2);
    unsigned short* vtb = (unsigned short*)carve((size_t)32 * 64 * 2048 * 2);
    unsigned short* attn = xb;  // xb dead after qkv_gemm; reuse for attention output

    const int NX = 4096 * 1024;
    pack_x_kernel<<<(NX + 255) / 256, 256, 0, stream>>>(x, xb, NX);
    transpose_w_kernel<<<(3072 * 1024 + 255) / 256, 256, 0, stream>>>(w_qkv, wqT, 1024, 3072);
    transpose_w_kernel<<<(1024 * 1024 + 255) / 256, 256, 0, stream>>>(w_out, woT, 1024, 1024);

    qkv_gemm_kernel<<<dim3(32, 48), 256, 0, stream>>>(xb, wqT, qb, kb, vtb);
    flash_attn_kernel<<<dim3(16, 32), 256, 0, stream>>>(qb, kb, vtb, attn);
    out_gemm_kernel<<<dim3(32, 16), 256, 0, stream>>>(attn, woT, out);
}
